// Deformation_82738249990712
// MI455X (gfx1250) — matrix operand order repaired and checked by reading, not compiled, not run on hardware
//
#include <hip/hip_runtime.h>
#include <hip/hip_bf16.h>

// ---------------------------------------------------------------------------
// MI455X (gfx1250) deep-snake deformation pipeline.
// GEMM-shaped ops -> v_wmma_f32_16x16x32_bf16 (bf16 in, f32 acc).
// Each wave owns a 32(o) x 32(col) register tile: 4 accumulators fed by
// 2 A-frags + 2 B-frags per K-step (2x operand reuse vs a single 16x16 tile).
// Activations are channel-last bf16 so all fragment loads are b128 pairs.
// Conv LDS window staged via GLOBAL_LOAD_ASYNC_TO_LDS_B128 (ASYNCcnt) when
// the toolchain exposes the builtin.
// ---------------------------------------------------------------------------

typedef __attribute__((ext_vector_type(16))) __bf16 v16bf;
typedef __attribute__((ext_vector_type(8)))  __bf16 v8bf;
typedef __attribute__((ext_vector_type(8)))  float  v8f;
typedef __attribute__((ext_vector_type(4)))  int    v4i;

typedef v4i __attribute__((address_space(1)))* gas_v4i_ptr;   // global (__device__)
typedef v4i __attribute__((address_space(3)))* las_v4i_ptr;   // LDS (__shared__)

#define NB    16
#define NPTS  1024
#define BNCOL (NB * NPTS)   // 16384 columns

#define WMMA_BF16(A, B, C) \
    __builtin_amdgcn_wmma_f32_16x16x32_bf16(false, (A), false, (B), (short)0, (C), false, false)

#if __has_builtin(__builtin_amdgcn_global_load_async_to_lds_b128)
#define HAVE_ASYNC_LDS 1
#else
#define HAVE_ASYNC_LDS 0
#endif

// ---------------------------------------------------------------------------
// Weight pre-pack into WMMA A-fragment order.
// A-matrix 16x32 bf16 layout (contiguous, matches B): lane L holds row M = L%16;
//   element e -> k = 16*(L/16) + e
// dst index = ((((ot*T + t)*KC + kc)*32 + lane)*16 + e)
// src is f32 weight of shape (O, Ctot, T); channels [c_off, c_off+Creal) used.
// ---------------------------------------------------------------------------
__global__ void k_pack_w(const float* __restrict__ W, __bf16* __restrict__ dst,
                         int T, int KC, int Ctot, int c_off, int Creal, int total)
{
    int idx = blockIdx.x * 256 + threadIdx.x;
    if (idx >= total) return;
    int e    = idx & 15;
    int lane = (idx >> 4) & 31;
    int kc   = (idx >> 9) % KC;
    int rem  = (idx >> 9) / KC;
    int t    = rem % T;
    int ot   = rem / T;
    int M    = ot * 16 + (lane & 15);
    int half = lane >> 4;
    int k    = 16 * half + e;
    int c    = kc * 32 + k;
    float v  = 0.0f;
    if (c < Creal) v = W[((size_t)M * Ctot + c_off + c) * T + t];
    dst[idx] = (__bf16)v;
}

// BatchNorm (eval) folding: inv = g/sqrt(v+eps), beta' = be - m*inv
__global__ void k_bnprep(const float* __restrict__ g, const float* __restrict__ be,
                         const float* __restrict__ m, const float* __restrict__ v,
                         float* __restrict__ inv, float* __restrict__ beta)
{
    int c = threadIdx.x;                      // 128 threads
    float iv = g[c] * rsqrtf(v[c] + 1e-5f);
    inv[c]  = iv;
    beta[c] = be[c] - m[c] * iv;
}

__global__ void k_copy_f32(const float* __restrict__ s, float* __restrict__ d, int n)
{
    int i = blockIdx.x * 256 + threadIdx.x;
    if (i < n) d[i] = s[i];
}

// ---------------------------------------------------------------------------
// Per-batch coordinate min + write loc channels (64,65) of head input, zero pad.
// ---------------------------------------------------------------------------
__global__ void k_vertex_prep(const float* __restrict__ verts, __bf16* __restrict__ Hin)
{
    __shared__ float sx[256], sy[256];
    int b = blockIdx.x, tid = threadIdx.x;
    float mx = 1e30f, my = 1e30f;
    for (int n = tid; n < NPTS; n += 256) {
        mx = fminf(mx, verts[(b * NPTS + n) * 2 + 0]);
        my = fminf(my, verts[(b * NPTS + n) * 2 + 1]);
    }
    sx[tid] = mx; sy[tid] = my;
    __syncthreads();
    for (int s = 128; s > 0; s >>= 1) {
        if (tid < s) {
            sx[tid] = fminf(sx[tid], sx[tid + s]);
            sy[tid] = fminf(sy[tid], sy[tid + s]);
        }
        __syncthreads();
    }
    float minx = sx[0], miny = sy[0];
    for (int n = tid; n < NPTS; n += 256) {
        int col = b * NPTS + n;
        float lx = (verts[col * 2 + 0] - minx) * 0.25f;   // rel / MOVE / 4
        float ly = (verts[col * 2 + 1] - miny) * 0.25f;
        __bf16* row = Hin + (size_t)col * 96;
        row[64] = (__bf16)lx;
        row[65] = (__bf16)ly;
        for (int c = 66; c < 96; ++c) row[c] = (__bf16)0.0f;
    }
}

// ---------------------------------------------------------------------------
// grid_sample (bilinear, zeros padding, align_corners=False).
// One block per column; statically unrolled per pyramid level so all sampling
// state stays in registers (no runtime-indexed arrays -> no scratch).
// ---------------------------------------------------------------------------
template<int W, int C>
__device__ __forceinline__ void sample_level(const float* __restrict__ f, int b,
                                             float vx, float vy,
                                             __bf16* __restrict__ out, int tid)
{
    float Wl = (float)W;
    float gx = vx * (Wl / 512.0f) - 0.5f;     // ((v/512)*2-1 +1)*W/2 - 0.5
    float gy = vy * (Wl / 512.0f) - 0.5f;
    float x0 = floorf(gx), y0 = floorf(gy);
    float wx1 = gx - x0, wx0 = 1.0f - wx1;
    float wy1 = gy - y0, wy0 = 1.0f - wy1;
    float lim = Wl - 1.0f;
    float x1 = x0 + 1.0f, y1 = y0 + 1.0f;
    float mx0 = (x0 >= 0.0f && x0 <= lim) ? 1.0f : 0.0f;
    float mx1 = (x1 >= 0.0f && x1 <= lim) ? 1.0f : 0.0f;
    float my0 = (y0 >= 0.0f && y0 <= lim) ? 1.0f : 0.0f;
    float my1 = (y1 >= 0.0f && y1 <= lim) ? 1.0f : 0.0f;
    int xi0 = (int)fminf(fmaxf(x0, 0.0f), lim);
    int xi1 = (int)fminf(fmaxf(x1, 0.0f), lim);
    int yi0 = (int)fminf(fmaxf(y0, 0.0f), lim);
    int yi1 = (int)fminf(fmaxf(y1, 0.0f), lim);
    int i00 = yi0 * W + xi0;  float w00 = wx0 * wy0 * mx0 * my0;
    int i01 = yi0 * W + xi1;  float w01 = wx1 * wy0 * mx1 * my0;
    int i10 = yi1 * W + xi0;  float w10 = wx0 * wy1 * mx0 * my1;
    int i11 = yi1 * W + xi1;  float w11 = wx1 * wy1 * mx1 * my1;
    const float* base = f + (size_t)b * C * W * W;
    for (int c = tid; c < C; c += 256) {
        const float* p = base + (size_t)c * W * W;
        float r = p[i00] * w00 + p[i01] * w01 + p[i10] * w10 + p[i11] * w11;
        out[c] = (__bf16)r;
    }
}

__global__ void k_grid_sample(const float* __restrict__ f0, const float* __restrict__ f1,
                              const float* __restrict__ f2, const float* __restrict__ f3,
                              const float* __restrict__ verts, __bf16* __restrict__ PF)
{
    int col = blockIdx.x;
    int b   = col >> 10;
    float vx = verts[col * 2 + 0];
    float vy = verts[col * 2 + 1];
    __bf16* out = PF + (size_t)col * 960;
    int tid = threadIdx.x;
    sample_level<128,  64>(f0, b, vx, vy, out,       tid);
    sample_level< 64, 128>(f1, b, vx, vy, out + 64,  tid);
    sample_level< 32, 256>(f2, b, vx, vy, out + 192, tid);
    sample_level< 16, 512>(f3, b, vx, vy, out + 448, tid);
}

// ---------------------------------------------------------------------------
// Circular dilated conv1d (k=9) as WMMA GEMM.
// Workgroup = (batch, 64-column tile); 8 waves, each computing a 32(o)x32(col)
// register tile (waves: og = w&3 over 128 outputs, cg = w>>2 over 2 col halves).
// X window (Cpad x (64+8*dil)) staged once in LDS (max 24 KB) via the CDNA5
// async Global->LDS pipe (ASYNCcnt) when available.
// Epilogue: +bias -> ReLU -> BN -> (+residual) -> bf16 channel-last store.
// ---------------------------------------------------------------------------
__global__ __launch_bounds__(256)
void k_conv9(const __bf16* __restrict__ X, int x_stride, int x_choff,
             const __bf16* __restrict__ Ap, const float* __restrict__ bias,
             const float* __restrict__ bninv, const float* __restrict__ bnbeta,
             __bf16* __restrict__ Y, int y_stride, int y_choff,
             int Cpad, int KC, int dil, int resid)
{
    __shared__ __bf16 xs[96 * 128];               // max window 96 x 128 ch = 24 KB
    int blk   = blockIdx.x;
    int b     = blk >> 4;                         // 16 column tiles of 64 per batch
    int ntile = blk & 15;
    int n0    = ntile * 64;
    int window = 64 + 8 * dil;
    int tid   = threadIdx.x;

    // circular-window fill, 16B per transfer
    int total8 = (window * Cpad) >> 3;
    for (int i = tid; i < total8; i += 256) {
        int elem = i << 3;
        int pos  = elem / Cpad;
        int c    = elem - pos * Cpad;
        int n    = (n0 - 4 * dil + pos + NPTS) & (NPTS - 1);
        const __bf16* g = X + (size_t)(b * NPTS + n) * x_stride + x_choff + c;
        __bf16* l = xs + pos * Cpad + c;
#if HAVE_ASYNC_LDS
        __builtin_amdgcn_global_load_async_to_lds_b128(
            (gas_v4i_ptr)(__bf16*)g, (las_v4i_ptr)l, 0, 0);
#else
        *(float4*)l = *(const float4*)g;
#endif
    }
#if HAVE_ASYNC_LDS
    asm volatile("s_wait_asynccnt 0x0" ::: "memory");
#endif
    __syncthreads();

    int wave = tid >> 5, lane = tid & 31;
    int og = wave & 3, cg = wave >> 2;
    int cl = lane & 15, half = lane >> 4;

    v8f acc00 = {0.f,0.f,0.f,0.f,0.f,0.f,0.f,0.f};
    v8f acc01 = acc00, acc10 = acc00, acc11 = acc00;
    const size_t tile_sz = (size_t)9 * KC * 512;
    const __bf16* Apt0 = Ap + (size_t)(og * 2) * tile_sz;
    const __bf16* Apt1 = Apt0 + tile_sz;
    int posb = cg * 32 + cl;

    for (int t = 0; t < 9; ++t) {
        const __bf16* xrow0 = xs + (posb + t * dil) * Cpad + 16 * half;
        const __bf16* xrow1 = xrow0 + 16 * Cpad;
        for (int kc = 0; kc < KC; ++kc) {
            size_t aoff = (size_t)(t * KC + kc) * 512 + lane * 16;
            v16bf a0 = *(const v16bf*)(Apt0 + aoff);
            v16bf a1 = *(const v16bf*)(Apt1 + aoff);
            v16bf b0 = *(const v16bf*)(xrow0 + kc * 32);
            v16bf b1 = *(const v16bf*)(xrow1 + kc * 32);
            acc00 = WMMA_BF16(a0, b0, acc00);
            acc01 = WMMA_BF16(a0, b1, acc01);
            acc10 = WMMA_BF16(a1, b0, acc10);
            acc11 = WMMA_BF16(a1, b1, acc11);
        }
    }

    v8f accs[2][2] = { { acc00, acc01 }, { acc10, acc11 } };
#pragma unroll
    for (int i = 0; i < 2; ++i) {
#pragma unroll
        for (int j = 0; j < 2; ++j) {
            int obase = og * 32 + i * 16 + 8 * half;
            int col   = b * NPTS + n0 + cg * 32 + j * 16 + cl;
            v8bf rin = {};
            if (resid) rin = *(const v8bf*)(X + (size_t)col * x_stride + x_choff + obase);
            v8bf outv;
#pragma unroll
            for (int r = 0; r < 8; ++r) {
                int o = obase + r;
                float v = accs[i][j][r] + bias[o];
                v = fmaxf(v, 0.0f);               // ReLU before BN (reference)
                v = v * bninv[o] + bnbeta[o];
                if (resid) v += (float)rin[r];
                outv[r] = (__bf16)v;
            }
            *(v8bf*)(Y + (size_t)col * y_stride + y_choff + obase) = outv;
        }
    }
}

// ---------------------------------------------------------------------------
// 1x1 conv as WMMA GEMM; each wave owns a 32(o)x32(col) register tile, B operand
// read straight from channel-last bf16 global memory (b128 pairs per lane).
// ---------------------------------------------------------------------------
__global__ __launch_bounds__(256)
void k_gemm1x1(const __bf16* __restrict__ X, int x_stride,
               const __bf16* __restrict__ Ap, const float* __restrict__ bias,
               const float* __restrict__ colbias, int colbias_stride,
               __bf16* __restrict__ Y, int y_stride,
               int KC, int Otiles32, int relu)
{
    int gid  = blockIdx.x * 8 + (threadIdx.x >> 5);
    int lane = threadIdx.x & 31;
    int ot   = gid % Otiles32;                    // 32-output tiles
    int ct   = gid / Otiles32;                    // 32-column tiles
    int cl = lane & 15, half = lane >> 4;
    int colbase = ct * 32;

    const __bf16* xrow0 = X + (size_t)(colbase + cl) * x_stride + 16 * half;
    const __bf16* xrow1 = xrow0 + (size_t)16 * x_stride;
    const size_t tile_sz = (size_t)KC * 512;
    const __bf16* Apt0 = Ap + (size_t)(ot * 2) * tile_sz;
    const __bf16* Apt1 = Apt0 + tile_sz;

    v8f acc00 = {0.f,0.f,0.f,0.f,0.f,0.f,0.f,0.f};
    v8f acc01 = acc00, acc10 = acc00, acc11 = acc00;
    for (int kc = 0; kc < KC; ++kc) {
        size_t aoff = (size_t)kc * 512 + lane * 16;
        v16bf a0 = *(const v16bf*)(Apt0 + aoff);
        v16bf a1 = *(const v16bf*)(Apt1 + aoff);
        v16bf b0 = *(const v16bf*)(xrow0 + kc * 32);
        v16bf b1 = *(const v16bf*)(xrow1 + kc * 32);
        acc00 = WMMA_BF16(a0, b0, acc00);
        acc01 = WMMA_BF16(a0, b1, acc01);
        acc10 = WMMA_BF16(a1, b0, acc10);
        acc11 = WMMA_BF16(a1, b1, acc11);
    }

    v8f accs[2][2] = { { acc00, acc01 }, { acc10, acc11 } };
#pragma unroll
    for (int i = 0; i < 2; ++i) {
#pragma unroll
        for (int j = 0; j < 2; ++j) {
            int obase = ot * 32 + i * 16 + 8 * half;
            int col   = colbase + j * 16 + cl;
            int b     = col >> 10;
            v8bf outv;
#pragma unroll
            for (int r = 0; r < 8; ++r) {
                int o = obase + r;
                float v = accs[i][j][r];
                if (bias)    v += bias[o];
                if (colbias) v += colbias[b * colbias_stride + o];
                if (relu)    v  = fmaxf(v, 0.0f);
                outv[r] = (__bf16)v;
            }
            *(v8bf*)(Y + (size_t)col * y_stride + obase) = outv;
        }
    }
}

// global max over N per (b, o): g[b][256]
__global__ void k_maxreduce(const __bf16* __restrict__ F, float* __restrict__ g)
{
    __shared__ float s[256];
    int o = blockIdx.x & 255;
    int b = blockIdx.x >> 8;
    float m = -1e30f;
    for (int n = threadIdx.x; n < NPTS; n += 256)
        m = fmaxf(m, (float)F[(size_t)(b * NPTS + n) * 256 + o]);
    s[threadIdx.x] = m;
    __syncthreads();
    for (int st = 128; st > 0; st >>= 1) {
        if (threadIdx.x < st) s[threadIdx.x] = fmaxf(s[threadIdx.x], s[threadIdx.x + st]);
        __syncthreads();
    }
    if (threadIdx.x == 0) g[b * 256 + o] = s[0];
}

// pred0's broadcast-g contribution: yg[b][o] = sum_c W[o][c]*g[b][c] + bias[o]
__global__ void k_gemv_g(const float* __restrict__ W, const float* __restrict__ bias,
                         const float* __restrict__ g, float* __restrict__ yg)
{
    int b = blockIdx.x, o = threadIdx.x;          // 16 x 256
    float acc = bias[o];
    const float* w = W + (size_t)o * 1280;        // pred0.w is (256,1280)
    const float* gb = g + b * 256;
    for (int c = 0; c < 256; ++c) acc += w[c] * gb[c];
    yg[b * 256 + o] = acc;
}

// pred2 (64->2) + vertex update + output store
__global__ void k_pred2_update(const __bf16* __restrict__ H2, const float* __restrict__ W,
                               const float* __restrict__ bias,
                               float* __restrict__ verts, float* __restrict__ outp)
{
    int col = blockIdx.x * 256 + threadIdx.x;
    if (col >= BNCOL) return;
    const __bf16* h = H2 + (size_t)col * 64;
    float a0 = bias[0], a1 = bias[1];
    for (int c = 0; c < 64; ++c) {
        float hv = (float)h[c];
        a0 += W[c] * hv;
        a1 += W[64 + c] * hv;
    }
    float vx = verts[col * 2 + 0] + a0;
    float vy = verts[col * 2 + 1] + a1;
    verts[col * 2 + 0] = vx;  verts[col * 2 + 1] = vy;
    outp[col * 2 + 0]  = vx;  outp[col * 2 + 1]  = vy;
}

// ---------------------------------------------------------------------------
// Host orchestration
// ---------------------------------------------------------------------------
static const int DILS_H[7] = { 1, 1, 1, 2, 2, 4, 4 };

extern "C" void kernel_launch(void* const* d_in, const int* in_sizes, int n_in,
                              void* d_out, int out_size, void* d_ws, size_t ws_size,
                              hipStream_t stream)
{
    const float* f0 = (const float*)d_in[0];
    const float* f1 = (const float*)d_in[1];
    const float* f2 = (const float*)d_in[2];
    const float* f3 = (const float*)d_in[3];
    const float* verts_in = (const float*)d_in[4];
    // d_in[5]=width, d_in[6]=height (constant 512, folded into grid math)

    // ---- param leaves (JAX pytree order: sorted dict keys) ----
    const float* fuse_b = (const float*)d_in[7];
    const float* fuse_w = (const float*)d_in[8];
    struct SnakeP {
        const float *fusion_b, *fusion_w;
        const float *head_b, *head_be, *head_g, *head_m, *head_v, *head_w;
        const float *pred0_b, *pred0_w, *pred1_b, *pred1_w, *pred2_b, *pred2_w;
        const float *res_b[7], *res_be[7], *res_g[7], *res_m[7], *res_v[7], *res_w[7];
    } sk[3];
    {
        int i = 9;
        for (int s = 0; s < 3; ++s) {
            sk[s].fusion_b = (const float*)d_in[i++]; sk[s].fusion_w = (const float*)d_in[i++];
            sk[s].head_b = (const float*)d_in[i++]; sk[s].head_be = (const float*)d_in[i++];
            sk[s].head_g = (const float*)d_in[i++]; sk[s].head_m  = (const float*)d_in[i++];
            sk[s].head_v = (const float*)d_in[i++]; sk[s].head_w  = (const float*)d_in[i++];
            sk[s].pred0_b = (const float*)d_in[i++]; sk[s].pred0_w = (const float*)d_in[i++];
            sk[s].pred1_b = (const float*)d_in[i++]; sk[s].pred1_w = (const float*)d_in[i++];
            sk[s].pred2_b = (const float*)d_in[i++]; sk[s].pred2_w = (const float*)d_in[i++];
            for (int r = 0; r < 7; ++r) {
                sk[s].res_b[r]  = (const float*)d_in[i++];
                sk[s].res_be[r] = (const float*)d_in[i++];
                sk[s].res_g[r]  = (const float*)d_in[i++];
                sk[s].res_m[r]  = (const float*)d_in[i++];
                sk[s].res_v[r]  = (const float*)d_in[i++];
                sk[s].res_w[r]  = (const float*)d_in[i++];
            }
        }
    }

    // ---- workspace carve-up ----
    char* ws = (char*)d_ws;
    size_t off = 0;
    auto alloc = [&](size_t bytes) -> void* {
        void* p = ws + off;
        off = (off + bytes + 255) & ~(size_t)255;
        return p;
    };
    float*  verts = (float*) alloc((size_t)BNCOL * 2 * 4);
    __bf16* PF    = (__bf16*)alloc((size_t)BNCOL * 960 * 2);
    __bf16* Hin   = (__bf16*)alloc((size_t)BNCOL * 96 * 2);
    __bf16* S     = (__bf16*)alloc((size_t)BNCOL * 1024 * 2);
    __bf16* F     = (__bf16*)alloc((size_t)BNCOL * 256 * 2);
    __bf16* H1    = (__bf16*)alloc((size_t)BNCOL * 256 * 2);
    __bf16* H2    = (__bf16*)alloc((size_t)BNCOL * 64 * 2);
    float*  gbuf  = (float*) alloc(NB * 256 * 4);
    float*  ygbuf = (float*) alloc(NB * 256 * 4);

    const int TOT_FUSE   = 4  * 1 * 30 * 512;   // O=64,  K=960
    const int TOT_HEAD   = 8  * 9 * 3  * 512;   // O=128, Cpad=96
    const int TOT_RES    = 8  * 9 * 4  * 512;   // O=128, C=128
    const int TOT_FUSION = 16 * 1 * 32 * 512;   // O=256, K=1024
    const int TOT_PRED0  = 16 * 1 * 32 * 512;   // O=256, K=1024 (state part)
    const int TOT_PRED1  = 4  * 1 * 8  * 512;   // O=64,  K=256
    __bf16* pFuse   = (__bf16*)alloc((size_t)TOT_FUSE * 2);
    __bf16* pHead   = (__bf16*)alloc((size_t)TOT_HEAD * 2);
    __bf16* pRes[7];
    for (int r = 0; r < 7; ++r) pRes[r] = (__bf16*)alloc((size_t)TOT_RES * 2);
    __bf16* pFusion = (__bf16*)alloc((size_t)TOT_FUSION * 2);
    __bf16* pPred0  = (__bf16*)alloc((size_t)TOT_PRED0 * 2);
    __bf16* pPred1  = (__bf16*)alloc((size_t)TOT_PRED1 * 2);
    float*  bnInv   = (float*)alloc(8 * 128 * 4);   // head + 7 res
    float*  bnBeta  = (float*)alloc(8 * 128 * 4);

    auto cdiv = [](int a, int b) { return (a + b - 1) / b; };

    // current vertices <- input vertices
    k_copy_f32<<<cdiv(BNCOL * 2, 256), 256, 0, stream>>>(verts_in, verts, BNCOL * 2);

    // fuse weights: shared across steps, pack once
    k_pack_w<<<cdiv(TOT_FUSE, 256), 256, 0, stream>>>(fuse_w, pFuse, 1, 30, 960, 0, 960, TOT_FUSE);

    for (int s = 0; s < 3; ++s) {
        const SnakeP& P = sk[s];

        // ---- pack this snake's weights + fold BN ----
        k_pack_w<<<cdiv(TOT_HEAD, 256), 256, 0, stream>>>(P.head_w, pHead, 9, 3, 66, 0, 66, TOT_HEAD);
        for (int r = 0; r < 7; ++r)
            k_pack_w<<<cdiv(TOT_RES, 256), 256, 0, stream>>>(P.res_w[r], pRes[r], 9, 4, 128, 0, 128, TOT_RES);
        k_pack_w<<<cdiv(TOT_FUSION, 256), 256, 0, stream>>>(P.fusion_w, pFusion, 1, 32, 1024, 0, 1024, TOT_FUSION);
        k_pack_w<<<cdiv(TOT_PRED0, 256), 256, 0, stream>>>(P.pred0_w, pPred0, 1, 32, 1280, 256, 1024, TOT_PRED0);
        k_pack_w<<<cdiv(TOT_PRED1, 256), 256, 0, stream>>>(P.pred1_w, pPred1, 1, 8, 256, 0, 256, TOT_PRED1);
        k_bnprep<<<1, 128, 0, stream>>>(P.head_g, P.head_be, P.head_m, P.head_v, bnInv, bnBeta);
        for (int r = 0; r < 7; ++r)
            k_bnprep<<<1, 128, 0, stream>>>(P.res_g[r], P.res_be[r], P.res_m[r], P.res_v[r],
                                            bnInv + (r + 1) * 128, bnBeta + (r + 1) * 128);

        // ---- per-batch min + loc channels of head input ----
        k_vertex_prep<<<NB, 256, 0, stream>>>(verts, Hin);

        // ---- pyramid sampling -> PF [BN][960] bf16 ----
        k_grid_sample<<<BNCOL, 256, 0, stream>>>(f0, f1, f2, f3, verts, PF);

        // ---- fuse 960->64 -> Hin channels [0,64) ----
        k_gemm1x1<<<(2 * 512) / 8, 256, 0, stream>>>(PF, 960, pFuse, fuse_b,
                                                     nullptr, 0, Hin, 96, 30, 2, 0);

        // ---- head conv (66->128, dil=1) -> S[:,0:128] ----
        k_conv9<<<NB * 16, 256, 0, stream>>>(Hin, 96, 0, pHead, P.head_b,
                                             bnInv, bnBeta, S, 1024, 0, 96, 3, 1, 0);

        // ---- 7 residual conv blocks -> S[:, (i+1)*128 : (i+2)*128] ----
        for (int r = 0; r < 7; ++r)
            k_conv9<<<NB * 16, 256, 0, stream>>>(S, 1024, r * 128, pRes[r], P.res_b[r],
                                                 bnInv + (r + 1) * 128, bnBeta + (r + 1) * 128,
                                                 S, 1024, (r + 1) * 128, 128, 4, DILS_H[r], 1);

        // ---- fusion 1024->256, then global max over N ----
        k_gemm1x1<<<(8 * 512) / 8, 256, 0, stream>>>(S, 1024, pFusion, P.fusion_b,
                                                     nullptr, 0, F, 256, 32, 8, 0);
        k_maxreduce<<<NB * 256, 256, 0, stream>>>(F, gbuf);

        // ---- pred0 = relu(Wg*g + Ws*state + b): g-part as per-(b,o) bias ----
        k_gemv_g<<<NB, 256, 0, stream>>>(P.pred0_w, P.pred0_b, gbuf, ygbuf);
        k_gemm1x1<<<(8 * 512) / 8, 256, 0, stream>>>(S, 1024, pPred0, nullptr,
                                                     ygbuf, 256, H1, 256, 32, 8, 1);

        // ---- pred1 256->64 + relu ----
        k_gemm1x1<<<(2 * 512) / 8, 256, 0, stream>>>(H1, 256, pPred1, P.pred1_b,
                                                     nullptr, 0, H2, 64, 8, 2, 1);

        // ---- pred2 64->2, vertex update, emit pred s ----
        k_pred2_update<<<cdiv(BNCOL, 256), 256, 0, stream>>>(H2, P.pred2_w, P.pred2_b,
                                                             verts,
                                                             (float*)d_out + (size_t)s * BNCOL * 2);
    }
}
